// VRNN_29351806501211
// MI455X (gfx1250) — matrix operand order repaired and checked by reading, not compiled, not run on hardware
//
#include <hip/hip_runtime.h>
#include <hip/hip_bf16.h>
#include <stdint.h>

// ---------------- problem constants ----------------
#define T_SEQ 628
#define BATCH 128
#define DH    512
#define HR    1200
#define HRP   1216   // HR padded to multiple of 32 for WMMA K
#define G4    4800   // 4*H_RNN

// ---------------- WMMA types ----------------
typedef __attribute__((ext_vector_type(16))) __bf16 v16bf;
typedef __attribute__((ext_vector_type(8)))  float  v8f;

union BfPack { uint32_t u[8]; v16bf v; };

__device__ inline uint32_t f2bf(float f) {
  uint32_t x = __float_as_uint(f);
  return (x + 0x7FFFu + ((x >> 16) & 1u)) >> 16;   // RNE fp32->bf16
}
__device__ inline uint32_t pack_bf2(float lo, float hi) {
  return f2bf(lo) | (f2bf(hi) << 16);
}

// A-matrix 16x32 bf16 tile (ISA 7.12.2): lane l: m = l&15, half = l>>4.
// VGPR v in 0..3 -> K = half*8 + 2v ; v in 4..7 -> K = 16 + half*8 + 2(v-4).
// A is bf16 row-major -> each lane: two contiguous 8-element runs = 2x b128.
__device__ inline v16bf load_a16(const uint16_t* __restrict__ A, int lda,
                                 int m0, int k0, int lane) {
  int m = m0 + (lane & 15);
  int half = lane >> 4;
  const uint16_t* p = A + (size_t)m * lda + k0 + half * 8;
  uint4 a = *(const uint4*)(p);        // K = half*8 + 0..7
  uint4 b = *(const uint4*)(p + 16);   // K = 16 + half*8 + 0..7
  BfPack pk;
  pk.u[0] = a.x; pk.u[1] = a.y; pk.u[2] = a.z; pk.u[3] = a.w;
  pk.u[4] = b.x; pk.u[5] = b.y; pk.u[6] = b.z; pk.u[7] = b.w;
  return pk.v;
}

// B-matrix 32x16 bf16 tile, pre-swizzled: tile (kt,nt) = 32 lanes * 8 u32.
__device__ inline v16bf load_b16(const uint32_t* __restrict__ B, int kt,
                                 int Ntiles, int nt, int lane) {
  const uint32_t* p = B + (((size_t)kt * Ntiles + nt) * 256 + lane * 8);
  uint4 a = *(const uint4*)(p);
  uint4 b = *(const uint4*)(p + 4);
  BfPack pk;
  pk.u[0] = a.x; pk.u[1] = a.y; pk.u[2] = a.z; pk.u[3] = a.w;
  pk.u[4] = b.x; pk.u[5] = b.y; pk.u[6] = b.z; pk.u[7] = b.w;
  return pk.v;
}

// ---------------- weight swizzle fp32 -> bf16 WMMA-B layout (one-time) ----
__global__ void k_swizzle(const float* __restrict__ W, int rowOff, int Ksrc,
                          int N, uint32_t* __restrict__ dst, int total) {
  int id = blockIdx.x * 256 + threadIdx.x;
  if (id >= total) return;
  int v    = id & 7;
  int lane = (id >> 3) & 31;
  int tile = id >> 8;
  int Ntiles = N >> 4;
  int kt = tile / Ntiles, nt = tile - kt * Ntiles;
  int half = lane >> 4;
  int n = nt * 16 + (lane & 15);
  int k = kt * 32 + (v < 4 ? half * 8 + 2 * v : 16 + half * 8 + 2 * (v - 4));
  float f0 = (k     < Ksrc) ? W[(size_t)(rowOff + k)     * N + n] : 0.f;
  float f1 = (k + 1 < Ksrc) ? W[(size_t)(rowOff + k + 1) * N + n] : 0.f;
  dst[id] = pack_bf2(f0, f1);
}

// ---------------- generic 3-input WMMA GEMM, M-rep x2 ----------------
// C[M=128,N] = act(A1*B1 + A2*B2 + A3*B3 + bias(+bias2)); A bf16 row-major.
// grid = (N/64, 128/32), block = 128. Each wave: one nt, two m-tiles
// (m0, m0+16) sharing every B tile -> 2 WMMAs per B load.
__global__ __launch_bounds__(128) void k_gemm(
    const uint16_t* A1, int lda1, int kt1, const uint32_t* B1,
    const uint16_t* A2, int lda2, int kt2, const uint32_t* B2,
    const uint16_t* A3, int lda3, int kt3, const uint32_t* B3,
    const float* __restrict__ bias, const float* __restrict__ bias2,
    float* C32, uint16_t* C16, int N, int relu) {
  int lane = threadIdx.x & 31;
  int nt   = blockIdx.x * 4 + (threadIdx.x >> 5);
  int Ntiles = N >> 4;
  int m0 = blockIdx.y * 32;
  v8f acc0 = {};
  v8f acc1 = {};
  for (int kt = 0; kt < kt1; ++kt) {
    v16bf b  = load_b16(B1, kt, Ntiles, nt, lane);
    v16bf a0 = load_a16(A1, lda1, m0,      kt * 32, lane);
    v16bf a1 = load_a16(A1, lda1, m0 + 16, kt * 32, lane);
    acc0 = __builtin_amdgcn_wmma_f32_16x16x32_bf16(false, a0, false, b, (short)0, acc0, false, false);
    acc1 = __builtin_amdgcn_wmma_f32_16x16x32_bf16(false, a1, false, b, (short)0, acc1, false, false);
  }
  if (A2) {
    for (int kt = 0; kt < kt2; ++kt) {
      v16bf b  = load_b16(B2, kt, Ntiles, nt, lane);
      v16bf a0 = load_a16(A2, lda2, m0,      kt * 32, lane);
      v16bf a1 = load_a16(A2, lda2, m0 + 16, kt * 32, lane);
      acc0 = __builtin_amdgcn_wmma_f32_16x16x32_bf16(false, a0, false, b, (short)0, acc0, false, false);
      acc1 = __builtin_amdgcn_wmma_f32_16x16x32_bf16(false, a1, false, b, (short)0, acc1, false, false);
    }
  }
  if (A3) {
    for (int kt = 0; kt < kt3; ++kt) {
      v16bf b  = load_b16(B3, kt, Ntiles, nt, lane);
      v16bf a0 = load_a16(A3, lda3, m0,      kt * 32, lane);
      v16bf a1 = load_a16(A3, lda3, m0 + 16, kt * 32, lane);
      acc0 = __builtin_amdgcn_wmma_f32_16x16x32_bf16(false, a0, false, b, (short)0, acc0, false, false);
      acc1 = __builtin_amdgcn_wmma_f32_16x16x32_bf16(false, a1, false, b, (short)0, acc1, false, false);
    }
  }
  // C/D layout: m = m0 + v + 8*half, n = nt*16 + (lane&15)
  int n = nt * 16 + (lane & 15);
  float bv = bias[n] + (bias2 ? bias2[n] : 0.f);
  int half = lane >> 4;
#pragma unroll
  for (int v = 0; v < 8; ++v) {
    int ma = m0 + v + 8 * half;
    float o0 = acc0[v] + bv;
    float o1 = acc1[v] + bv;
    if (relu) { o0 = fmaxf(o0, 0.f); o1 = fmaxf(o1, 0.f); }
    if (C32) { C32[(size_t)ma * N + n] = o0; C32[(size_t)(ma + 16) * N + n] = o1; }
    if (C16) {
      C16[(size_t)ma * N + n]        = (uint16_t)f2bf(o0);
      C16[(size_t)(ma + 16) * N + n] = (uint16_t)f2bf(o1);
    }
  }
}

// ---------------- small kernels ----------------
__global__ void k_zero32(float* p, int n) {
  int i = blockIdx.x * 256 + threadIdx.x;
  if (i < n) p[i] = 0.f;
}
__global__ void k_zero16(uint16_t* p, int n) {
  int i = blockIdx.x * 256 + threadIdx.x;
  if (i < n) p[i] = 0;
}

// fx layer0: relu(x_t @ fx_w0 + b0), K=3 -> bf16 out
__global__ void k_fx0(const float* __restrict__ x, const float* __restrict__ w,
                      const float* __restrict__ b, uint16_t* __restrict__ out, int t) {
  int id = blockIdx.x * 256 + threadIdx.x;       // 128*512
  int bi = id >> 9, j = id & 511;
  const float* xr = x + (size_t)t * BATCH * 3 + bi * 3;
  float acc = b[j] + xr[0] * w[j] + xr[1] * w[512 + j] + xr[2] * w[1024 + j];
  out[id] = (uint16_t)f2bf(fmaxf(acc, 0.f));
}

// inf heads + reparameterize + fz layer0 (K=3) -> bf16 fz0
__global__ void k_mid(const float* __restrict__ enc2,
                      const float* __restrict__ wm, const float* __restrict__ bm,
                      const float* __restrict__ wl, const float* __restrict__ bl,
                      const float* __restrict__ noise,
                      const float* __restrict__ fzw0, const float* __restrict__ fzb0,
                      float* __restrict__ zm, float* __restrict__ zl,
                      uint16_t* __restrict__ fz0, int t) {
  int b = blockIdx.x, tid = threadIdx.x;          // block=512, grid=128
  __shared__ float sm[3], sl[3], sz[3];
  if (tid < 6) {
    int c = tid % 3; int isl = tid / 3;
    const float* w = isl ? wl : wm;
    float acc = isl ? bl[c] : bm[c];
    const float* e = enc2 + (size_t)b * DH;
    for (int k = 0; k < DH; ++k) acc += e[k] * w[k * 3 + c];
    size_t o = (size_t)t * BATCH * 3 + b * 3 + c;
    if (isl) { sl[c] = acc; zl[o] = acc; } else { sm[c] = acc; zm[o] = acc; }
  }
  __syncthreads();
  if (tid < 3)
    sz[tid] = sm[tid] + noise[(size_t)t * BATCH * 3 + b * 3 + tid] * expf(0.5f * sl[tid]);
  __syncthreads();
  float acc = fzb0[tid] + sz[0] * fzw0[tid] + sz[1] * fzw0[DH + tid] + sz[2] * fzw0[2 * DH + tid];
  fz0[(size_t)b * DH + tid] = (uint16_t)f2bf(fmaxf(acc, 0.f));
}

// y = dec2 @ gen_w + gen_b  (N=3)
__global__ void k_y(const float* __restrict__ dec2, const float* __restrict__ w,
                    const float* __restrict__ b, float* __restrict__ yseq, int t) {
  int id = blockIdx.x * 128 + threadIdx.x;
  if (id >= BATCH * 3) return;
  int bi = id / 3, c = id % 3;
  const float* r = dec2 + (size_t)bi * DH;
  float acc = b[c];
  for (int k = 0; k < DH; ++k) acc += r[k] * w[k * 3 + c];
  yseq[(size_t)t * BATCH * 3 + id] = acc;
}

// prior_mean / prior_logvar heads (N=3 each)
__global__ void k_pheads(const float* __restrict__ pr2,
                         const float* __restrict__ wm, const float* __restrict__ bm,
                         const float* __restrict__ wl, const float* __restrict__ bl,
                         float* __restrict__ zmp, float* __restrict__ zlp, int t) {
  int id = blockIdx.x * 128 + threadIdx.x;
  if (id >= 2 * BATCH * 3) return;
  int head = id >= BATCH * 3;
  int rem = id - head * BATCH * 3;
  int bi = rem / 3, c = rem % 3;
  const float* w = head ? wl : wm;
  float acc = head ? bl[c] : bm[c];
  const float* r = pr2 + (size_t)bi * DH;
  for (int k = 0; k < DH; ++k) acc += r[k] * w[k * 3 + c];
  (head ? zlp : zmp)[(size_t)t * BATCH * 3 + rem] = acc;
}

// LSTM pointwise; h written as bf16 (pad lanes kept zero), c stays fp32
__global__ void k_lstm(const float* __restrict__ gates, uint16_t* __restrict__ h,
                       float* __restrict__ c) {
  int id = blockIdx.x * 256 + threadIdx.x;        // BATCH*HRP
  if (id >= BATCH * HRP) return;
  int bi = id / HRP, j = id - bi * HRP;
  if (j >= HR) { h[id] = 0; return; }
  const float* g = gates + (size_t)bi * G4;
  float gi = g[j], gf = g[HR + j], gg = g[2 * HR + j], go = g[3 * HR + j];
  float si = 1.f / (1.f + expf(-gi));
  float sf = 1.f / (1.f + expf(-gf));
  float so = 1.f / (1.f + expf(-go));
  float cn = sf * c[(size_t)bi * HR + j] + si * tanhf(gg);
  float hn = so * tanhf(cn);
  c[(size_t)bi * HR + j] = cn;
  h[id] = (uint16_t)f2bf(hn);
}

// ---------------- loss reduction ----------------
__device__ float blkred256(float v, float* s) {
  int tid = threadIdx.x;
  s[tid] = v; __syncthreads();
  for (int o = 128; o > 0; o >>= 1) { if (tid < o) s[tid] += s[tid + o]; __syncthreads(); }
  float r = s[0]; __syncthreads(); return r;
}
__device__ float blkred512(float v, float* s) {
  int tid = threadIdx.x;
  s[tid] = v; __syncthreads();
  for (int o = 256; o > 0; o >>= 1) { if (tid < o) s[tid] += s[tid + o]; __syncthreads(); }
  float r = s[0]; __syncthreads(); return r;
}

__global__ void k_loss1(const float* __restrict__ x, const int* __restrict__ lengths,
                        const float* __restrict__ yseq,
                        const float* __restrict__ zm, const float* __restrict__ zl,
                        const float* __restrict__ zmp, const float* __restrict__ zlp,
                        float* __restrict__ partial) {
  __shared__ float s[256];
  int id = blockIdx.x * 256 + threadIdx.x;        // T*B = 80384 exactly
  int t = id >> 7, b = id & 127;
  float lb = 0.f, lg = 0.f, lk = 0.f;
  if (t < lengths[b]) {
    size_t o3 = (size_t)id * 3;
    float l = yseq[o3], tgt = x[o3];
    lb = fmaxf(l, 0.f) - l * tgt + log1pf(expf(-fabsf(l)));
    const float inv2s2 = 50.f;                       // 1/(2*0.1^2)
    const float cg = 0.5f * logf(2.f * 3.14159265358979323846f * 0.01f);
#pragma unroll
    for (int c = 1; c < 3; ++c) {
      float d = x[o3 + c] - yseq[o3 + c];
      lg += cg + d * d * inv2s2;
    }
#pragma unroll
    for (int c = 0; c < 3; ++c) {
      float a = zl[o3 + c], ap = zlp[o3 + c], m = zm[o3 + c], mp = zmp[o3 + c];
      lk += -0.5f * (1.f + a - ap - (expf(a) + (m - mp) * (m - mp)) * expf(-ap));
    }
  }
  float r0 = blkred256(lb, s);
  float r1 = blkred256(lg, s);
  float r2 = blkred256(lk, s);
  if (threadIdx.x == 0) {
    partial[blockIdx.x * 3 + 0] = r0;
    partial[blockIdx.x * 3 + 1] = r1;
    partial[blockIdx.x * 3 + 2] = r2;
  }
}

__global__ void k_loss2(const float* __restrict__ partial, int nblk,
                        const int* __restrict__ lengths, float* __restrict__ out) {
  __shared__ float s[512];
  int tid = threadIdx.x;
  float a0 = 0, a1 = 0, a2 = 0, dn = 0;
  for (int i = tid; i < nblk; i += 512) {
    a0 += partial[i * 3]; a1 += partial[i * 3 + 1]; a2 += partial[i * 3 + 2];
  }
  if (tid < BATCH) dn = (float)lengths[tid];
  float r0 = blkred512(a0, s);
  float r1 = blkred512(a1, s);
  float r2 = blkred512(a2, s);
  float rd = blkred512(dn, s);
  if (tid == 0) {
    float lb = r0 / rd, lg = r1 / rd, lk = r2 / rd;
    out[0] = lb + lg + lk; out[1] = lb; out[2] = lg; out[3] = lk;
  }
}

// ---------------- host side ----------------
extern "C" void kernel_launch(void* const* d_in, const int* in_sizes, int n_in,
                              void* d_out, int out_size, void* d_ws, size_t ws_size,
                              hipStream_t stream) {
  (void)in_sizes; (void)n_in; (void)out_size; (void)ws_size;
  const float* x       = (const float*)d_in[0];
  const int*   lengths = (const int*)  d_in[1];
  const float* noise   = (const float*)d_in[2];
  const float* fx_w0 = (const float*)d_in[3];  const float* fx_b0 = (const float*)d_in[4];
  const float* fx_w1 = (const float*)d_in[5];  const float* fx_b1 = (const float*)d_in[6];
  const float* fz_w0 = (const float*)d_in[7];  const float* fz_b0 = (const float*)d_in[8];
  const float* fz_w1 = (const float*)d_in[9];  const float* fz_b1 = (const float*)d_in[10];
  const float* hxz_w0 = (const float*)d_in[11]; const float* hxz_b0 = (const float*)d_in[12];
  const float* hxz_w1 = (const float*)d_in[13]; const float* hxz_b1 = (const float*)d_in[14];
  const float* hz_w0 = (const float*)d_in[15];  const float* hz_b0 = (const float*)d_in[16];
  const float* hz_w1 = (const float*)d_in[17];  const float* hz_b1 = (const float*)d_in[18];
  const float* hzx_w0 = (const float*)d_in[19]; const float* hzx_b0 = (const float*)d_in[20];
  const float* hzx_w1 = (const float*)d_in[21]; const float* hzx_b1 = (const float*)d_in[22];
  const float* im_w = (const float*)d_in[23];   const float* im_b = (const float*)d_in[24];
  const float* il_w = (const float*)d_in[25];   const float* il_b = (const float*)d_in[26];
  const float* pm_w = (const float*)d_in[27];   const float* pm_b = (const float*)d_in[28];
  const float* pl_w = (const float*)d_in[29];   const float* pl_b = (const float*)d_in[30];
  const float* gen_w = (const float*)d_in[31];  const float* gen_b = (const float*)d_in[32];
  const float* wih_w = (const float*)d_in[33];  const float* wih_b = (const float*)d_in[34];
  const float* whh_w = (const float*)d_in[35];  const float* whh_b = (const float*)d_in[36];

  // workspace carve (bytes, 256-aligned); total ~ 36 MB
  uint8_t* wsp = (uint8_t*)d_ws;
  auto carve = [&](size_t bytes) -> void* {
    void* p = wsp; wsp += (bytes + 255) & ~(size_t)255; return p;
  };
  uint32_t* Wfx1   = (uint32_t*)carve(131072u * 4);   // 512x512 bf16 /2
  uint32_t* Whxz0a = (uint32_t*)carve(131072u * 4);
  uint32_t* Whxz0b = (uint32_t*)carve(311296u * 4);   // 1216x512
  uint32_t* Whxz1  = (uint32_t*)carve(131072u * 4);
  uint32_t* Wfz1   = (uint32_t*)carve(131072u * 4);
  uint32_t* Whzx0a = (uint32_t*)carve(131072u * 4);
  uint32_t* Whzx0b = (uint32_t*)carve(311296u * 4);
  uint32_t* Whzx1  = (uint32_t*)carve(131072u * 4);
  uint32_t* Whz0   = (uint32_t*)carve(311296u * 4);
  uint32_t* Whz1   = (uint32_t*)carve(131072u * 4);
  uint32_t* Wwiha  = (uint32_t*)carve(1228800u * 4);  // 512x4800
  uint32_t* Wwihb  = (uint32_t*)carve(1228800u * 4);
  uint32_t* Wwhh   = (uint32_t*)carve(2918400u * 4);  // 1216x4800
  uint16_t* fx0buf = (uint16_t*)carve(65536u * 2);    // bf16 activations
  uint16_t* fxt    = (uint16_t*)carve(65536u * 2);
  uint16_t* enc    = (uint16_t*)carve(65536u * 2);
  uint16_t* fz0b   = (uint16_t*)carve(65536u * 2);
  uint16_t* fzt    = (uint16_t*)carve(65536u * 2);
  uint16_t* dec    = (uint16_t*)carve(65536u * 2);
  uint16_t* pr1    = (uint16_t*)carve(65536u * 2);
  uint16_t* hbuf   = (uint16_t*)carve((size_t)BATCH * HRP * 2);
  float* enc2   = (float*)carve(65536u * 4);          // fp32 for pointwise
  float* dec2   = (float*)carve(65536u * 4);
  float* pr2    = (float*)carve(65536u * 4);
  float* gates  = (float*)carve((size_t)BATCH * G4 * 4);
  float* cbuf   = (float*)carve((size_t)BATCH * HR * 4);
  float* zm_s   = (float*)carve((size_t)T_SEQ * BATCH * 3 * 4);
  float* zl_s   = (float*)carve((size_t)T_SEQ * BATCH * 3 * 4);
  float* zmp_s  = (float*)carve((size_t)T_SEQ * BATCH * 3 * 4);
  float* zlp_s  = (float*)carve((size_t)T_SEQ * BATCH * 3 * 4);
  float* y_s    = (float*)carve((size_t)T_SEQ * BATCH * 3 * 4);
  float* partial = (float*)carve(4096);

  auto swz = [&](const float* W, int rowOff, int Ksrc, int Kpad, int N, uint32_t* dst) {
    int total = (Kpad / 2) * N;
    k_swizzle<<<dim3((total + 255) / 256), dim3(256), 0, stream>>>(W, rowOff, Ksrc, N, dst, total);
  };
  swz(fx_w1, 0, 512, 512, 512, Wfx1);
  swz(hxz_w0, 0, 512, 512, 512, Whxz0a);
  swz(hxz_w0, 512, 1200, HRP, 512, Whxz0b);
  swz(hxz_w1, 0, 512, 512, 512, Whxz1);
  swz(fz_w1, 0, 512, 512, 512, Wfz1);
  swz(hzx_w0, 0, 512, 512, 512, Whzx0a);
  swz(hzx_w0, 512, 1200, HRP, 512, Whzx0b);
  swz(hzx_w1, 0, 512, 512, 512, Whzx1);
  swz(hz_w0, 0, 1200, HRP, 512, Whz0);
  swz(hz_w1, 0, 512, 512, 512, Whz1);
  swz(wih_w, 0, 512, 512, G4, Wwiha);
  swz(wih_w, 512, 512, 512, G4, Wwihb);
  swz(whh_w, 0, 1200, HRP, G4, Wwhh);

  k_zero16<<<dim3((BATCH * HRP + 255) / 256), dim3(256), 0, stream>>>(hbuf, BATCH * HRP);
  k_zero32<<<dim3((BATCH * HR + 255) / 256), dim3(256), 0, stream>>>(cbuf, BATCH * HR);

  auto gemm = [&](const uint16_t* A1, int lda1, int kt1, const uint32_t* B1,
                  const uint16_t* A2, int lda2, int kt2, const uint32_t* B2,
                  const uint16_t* A3, int lda3, int kt3, const uint32_t* B3,
                  const float* bias, const float* bias2,
                  float* C32, uint16_t* C16, int N, int relu) {
    k_gemm<<<dim3(N / 64, BATCH / 32), dim3(128), 0, stream>>>(
        A1, lda1, kt1, B1, A2, lda2, kt2, B2, A3, lda3, kt3, B3,
        bias, bias2, C32, C16, N, relu);
  };

  for (int t = 0; t < T_SEQ; ++t) {
    // feature_x for this step
    k_fx0<<<dim3(256), dim3(256), 0, stream>>>(x, fx_w0, fx_b0, fx0buf, t);
    gemm(fx0buf, 512, 16, Wfx1, nullptr, 0, 0, nullptr, nullptr, 0, 0, nullptr,
         fx_b1, nullptr, nullptr, fxt, 512, 1);
    // enc = relu(relu([fxt,h] @ hxz_w0 + b0) @ hxz_w1 + b1)
    gemm(fxt, 512, 16, Whxz0a, hbuf, HRP, HRP / 32, Whxz0b, nullptr, 0, 0, nullptr,
         hxz_b0, nullptr, nullptr, enc, 512, 1);
    gemm(enc, 512, 16, Whxz1, nullptr, 0, 0, nullptr, nullptr, 0, 0, nullptr,
         hxz_b1, nullptr, enc2, nullptr, 512, 1);
    // mean/logvar heads, reparam, fz layer0
    k_mid<<<dim3(BATCH), dim3(512), 0, stream>>>(enc2, im_w, im_b, il_w, il_b,
                                                 noise, fz_w0, fz_b0, zm_s, zl_s, fz0b, t);
    gemm(fz0b, 512, 16, Wfz1, nullptr, 0, 0, nullptr, nullptr, 0, 0, nullptr,
         fz_b1, nullptr, nullptr, fzt, 512, 1);
    // dec = relu(relu([fzt,h] @ hzx_w0 + b0) @ hzx_w1 + b1)
    gemm(fzt, 512, 16, Whzx0a, hbuf, HRP, HRP / 32, Whzx0b, nullptr, 0, 0, nullptr,
         hzx_b0, nullptr, nullptr, dec, 512, 1);
    gemm(dec, 512, 16, Whzx1, nullptr, 0, 0, nullptr, nullptr, 0, 0, nullptr,
         hzx_b1, nullptr, dec2, nullptr, 512, 1);
    k_y<<<dim3(3), dim3(128), 0, stream>>>(dec2, gen_w, gen_b, y_s, t);
    // gates = [fxt,fzt] @ wih + h @ whh + bih + bhh
    gemm(fxt, 512, 16, Wwiha, fzt, 512, 16, Wwihb, hbuf, HRP, HRP / 32, Wwhh,
         wih_b, whh_b, gates, nullptr, G4, 0);
    // prior MLP on carry-in h (== h_seq[t]) -- before LSTM update
    gemm(hbuf, HRP, HRP / 32, Whz0, nullptr, 0, 0, nullptr, nullptr, 0, 0, nullptr,
         hz_b0, nullptr, nullptr, pr1, 512, 1);
    gemm(pr1, 512, 16, Whz1, nullptr, 0, 0, nullptr, nullptr, 0, 0, nullptr,
         hz_b1, nullptr, pr2, nullptr, 512, 1);
    k_pheads<<<dim3(6), dim3(128), 0, stream>>>(pr2, pm_w, pm_b, pl_w, pl_b, zmp_s, zlp_s, t);
    // LSTM state update (writes h bf16 + c fp32; keeps h pad zeroed)
    k_lstm<<<dim3((BATCH * HRP + 255) / 256), dim3(256), 0, stream>>>(gates, hbuf, cbuf);
  }

  // masked loss reduction: 80384 = 314 * 256
  k_loss1<<<dim3(314), dim3(256), 0, stream>>>(x, lengths, y_s, zm_s, zl_s, zmp_s, zlp_s, partial);
  k_loss2<<<dim3(1), dim3(512), 0, stream>>>(partial, 314, lengths, (float*)d_out);
}
